// FFNAttention_19731079757918
// MI455X (gfx1250) — matrix operand order repaired and checked by reading, not compiled, not run on hardware
//
#include <hip/hip_runtime.h>
#include <hip/hip_bf16.h>

// Problem sizes (match reference)
constexpr int B = 32, S = 4096, I = 512, H = 512;
constexpr int Sc = 64;            // rows of S per chunk
constexpr int C  = S / Sc;        // 64 chunks per batch row
constexpr float NEG_INF = -1e18f;

typedef __attribute__((ext_vector_type(16))) _Float16 v16h;
typedef __attribute__((ext_vector_type(8)))  float    v8f;
typedef int async_v4i __attribute__((vector_size(16)));   // b128 payload type for async-LDS builtin

#if defined(__HIP_DEVICE_COMPILE__) && __has_builtin(__builtin_amdgcn_global_load_async_to_lds_b128)
#define HAVE_ASYNC_LDS 1
#else
#define HAVE_ASYNC_LDS 0
#endif

// ---------------------------------------------------------------------------
// Kernel A: w[i] = sum_h Wk[h,i]*Wout[h]  (block 0) ; u[i] = sum_h Wq[h,i]*Wout[h] (block 1)
__global__ void prep_wu(const float* __restrict__ Wk, const float* __restrict__ Wq,
                        const float* __restrict__ Wout,
                        float* __restrict__ wv, float* __restrict__ uv) {
  const float* Wsrc = blockIdx.x ? Wq : Wk;
  float* dst        = blockIdx.x ? uv : wv;
  int t = threadIdx.x;
  for (int i = t; i < I; i += 256) {
    float acc = 0.f;
    for (int h = 0; h < H; ++h) acc += Wsrc[h * I + i] * Wout[h];
    dst[i] = acc;
  }
}

// ---------------------------------------------------------------------------
// Kernel A2: base[b] = query[b]·u + (bq+bk)·Wout
__global__ void prep_base(const float* __restrict__ query, const float* __restrict__ uv,
                          const float* __restrict__ bq, const float* __restrict__ bk,
                          const float* __restrict__ Wout, float* __restrict__ bs) {
  __shared__ float red8[8];
  int b = blockIdx.x, t = threadIdx.x;
  float acc = 0.f;
  for (int i = t; i < I; i += 256)
    acc += query[b * I + i] * uv[i] + (bq[i] + bk[i]) * Wout[i];
  for (int off = 16; off > 0; off >>= 1) acc += __shfl_xor(acc, off, 32);
  if ((t & 31) == 0) red8[t >> 5] = acc;
  __syncthreads();
  if (t == 0) { float s = 0.f; for (int k = 0; k < 8; ++k) s += red8[k]; bs[b] = s; }
}

// ---------------------------------------------------------------------------
// Kernel B: single streaming pass over `key`. One block per (b, chunk j).
//  - stages the 64x512 f32 key tile in LDS (128 KB; gfx1250 allows 320 KB/WG)
//    via GLOBAL_LOAD_ASYNC_TO_LDS_B128 (ASYNCcnt path, no VGPR round-trip)
//  - scores[s] = key[s]·w + base[b], masked -> writes attn_weight
//  - chunk softmax partials: m_j, l_j, c_j[i] = sum_s exp(sc[s]-m_j)*key[s,i]
__global__ void fused_scores(const float* __restrict__ key, const int* __restrict__ mask,
                             const float* __restrict__ wv, const float* __restrict__ baseg,
                             float* __restrict__ attn_weight,
                             float* __restrict__ mp, float* __restrict__ lp,
                             float* __restrict__ cp) {
  extern __shared__ float smem[];
  float* tile = smem;              // 64*512 = 32768 floats
  float* ws_  = smem + 32768;      // 512
  float* sc   = smem + 33280;      // 64  (masked scores)
  float* esc  = smem + 33344;      // 64  (exp(sc - m))

  const int t = threadIdx.x;
  const int b = blockIdx.x / C;
  const int j = blockIdx.x % C;
  const long chunkBase = ((long)b * S + (long)j * Sc) * I;

  ws_[t] = wv[t];
  ws_[t + 256] = wv[t + 256];

  // Stream 128 KB of `key` into LDS (b128 granularity, 32 issues/thread)
  const float4* gsrc = (const float4*)(key + chunkBase);
  float4* tl4 = (float4*)tile;
#if HAVE_ASYNC_LDS
#pragma unroll
  for (int it = 0; it < 32; ++it) {
    __builtin_amdgcn_global_load_async_to_lds_b128(
        (__attribute__((address_space(1))) async_v4i*)(gsrc + it * 256 + t),
        (__attribute__((address_space(3))) async_v4i*)(tl4 + it * 256 + t),
        0, 0);
  }
  __builtin_amdgcn_s_wait_asynccnt(0);
#else
#pragma unroll
  for (int it = 0; it < 32; ++it)
    tl4[it * 256 + t] = gsrc[it * 256 + t];
#endif
  __syncthreads();

  // Scores: wave-per-row (8 waves x 8 rows), wave32 shuffle reduction
  const int wid = t >> 5, lane = t & 31;
  const float bb = baseg[b];
  for (int r = 0; r < 8; ++r) {
    int row = wid * 8 + r;
    const float* rp = tile + row * I;
    float acc = 0.f;
#pragma unroll
    for (int q4 = 0; q4 < 4; ++q4) {
      int i0 = q4 * 128 + lane * 4;
      float4 kv  = *(const float4*)(rp  + i0);
      float4 wv4 = *(const float4*)(ws_ + i0);
      acc += kv.x * wv4.x + kv.y * wv4.y + kv.z * wv4.z + kv.w * wv4.w;
    }
#pragma unroll
    for (int off = 16; off > 0; off >>= 1) acc += __shfl_xor(acc, off, 32);
    if (lane == 0) {
      int s = j * Sc + row;
      float aw = mask[b * S + s] ? NEG_INF : (acc + bb);
      sc[row] = aw;
      attn_weight[b * S + s] = aw;     // masked logits are an output
    }
  }
  __syncthreads();

  // Chunk softmax partials
  float m = -3.0e38f;
  for (int s = 0; s < Sc; ++s) m = fmaxf(m, sc[s]);
  if (t < Sc) esc[t] = __expf(sc[t] - m);
  __syncthreads();
  float l = 0.f;
  for (int s = 0; s < Sc; ++s) l += esc[s];

  // c_j[i]: each thread owns i = 2t, 2t+1 (conflict-free float2 LDS reads)
  float c0 = 0.f, c1 = 0.f;
  const float2* tl2 = (const float2*)tile;
  for (int s = 0; s < Sc; ++s) {
    float e = esc[s];
    float2 kv = tl2[s * (I / 2) + t];
    c0 += e * kv.x;
    c1 += e * kv.y;
  }
  int pidx = b * C + j;
  if (t == 0) { mp[pidx] = m; lp[pidx] = l; }
  float* cpd = cp + (long)pidx * I;
  cpd[2 * t]     = c0;
  cpd[2 * t + 1] = c1;
}

// ---------------------------------------------------------------------------
// Kernel C: merge chunk partials -> cfinal[b,i] = (1/L) sum_j exp(m_j - M) c_j[i]
__global__ void combine(const float* __restrict__ mp, const float* __restrict__ lp,
                        const float* __restrict__ cp, float* __restrict__ cf) {
  __shared__ float coef[C];
  __shared__ float Msh, Lsh;
  int b = blockIdx.x, t = threadIdx.x;
  if (t == 0) {
    float M = -3.0e38f;
    for (int jj = 0; jj < C; ++jj) M = fmaxf(M, mp[b * C + jj]);
    float L = 0.f;
    for (int jj = 0; jj < C; ++jj) L += lp[b * C + jj] * __expf(mp[b * C + jj] - M);
    Msh = M; Lsh = L;
  }
  __syncthreads();
  if (t < C) coef[t] = __expf(mp[b * C + t] - Msh) / Lsh;
  __syncthreads();
  for (int i = t; i < I; i += 256) {
    float acc = 0.f;
    for (int jj = 0; jj < C; ++jj) acc += coef[jj] * cp[((long)(b * C + jj)) * I + i];
    cf[b * I + i] = acc;
  }
}

// ---------------------------------------------------------------------------
// Kernel D: attn = cfinal @ Wk^T + bk via v_wmma_f32_16x16x32_f16.
// f32 operands are split hi/lo into f16 (3 WMMAs per k-step -> ~f32 accuracy).
// One wave per 16x16 output tile; 64 tiles total.
__global__ void attn_wmma(const float* __restrict__ cf, const float* __restrict__ Wk,
                          const float* __restrict__ bk, float* __restrict__ attn) {
  int tile = blockIdx.x;          // 0..63
  int m0 = (tile >> 5) * 16;      // batch-tile origin (B=32 -> 2 tiles)
  int n0 = (tile & 31) * 16;      // H-tile origin (512/16 = 32 tiles)
  int lane = threadIdx.x;         // wave32
  int row = lane & 15, hig = lane >> 4;

  v8f acc = {};
  const float* arow = cf + (m0 + row) * I;   // A[M=row, K]
  const float* brow = Wk + (n0 + row) * I;   // B[K, N=row] = Wk[n0+row, K]

  for (int kk = 0; kk < I; kk += 32) {
    v16h ah, al, bh, bl;
    // A 16x32 f16 layout: halves e=0..7  <- arow[kk + hig*16 + e]
    //                     halves e=8..15 <- arow[kk + hig*16 + 8 + (e-8)]
#pragma unroll
    for (int e = 0; e < 8; ++e) {
      float v0 = arow[kk + hig * 16 + e];
      _Float16 h0 = (_Float16)v0;
      ah[e] = h0; al[e] = (_Float16)(v0 - (float)h0);
      float v1 = arow[kk + hig * 16 + 8 + e];
      _Float16 h1 = (_Float16)v1;
      ah[e + 8] = h1; al[e + 8] = (_Float16)(v1 - (float)h1);
    }
    // B 32x16 f16 layout: K = e + hig*16, N = row
#pragma unroll
    for (int e = 0; e < 16; ++e) {
      float v = brow[kk + hig * 16 + e];
      _Float16 h = (_Float16)v;
      bh[e] = h; bl[e] = (_Float16)(v - (float)h);
    }
    acc = __builtin_amdgcn_wmma_f32_16x16x32_f16(false, ah, false, bh, (short)0, acc, false, false);
    acc = __builtin_amdgcn_wmma_f32_16x16x32_f16(false, ah, false, bl, (short)0, acc, false, false);
    acc = __builtin_amdgcn_wmma_f32_16x16x32_f16(false, al, false, bh, (short)0, acc, false, false);
  }

  // D layout: VGPR p -> M = p + hig*8, N = row
  float bias = bk[n0 + row];
#pragma unroll
  for (int p = 0; p < 8; ++p)
    attn[(m0 + p + hig * 8) * H + n0 + row] = acc[p] + bias;
}

// ---------------------------------------------------------------------------
extern "C" void kernel_launch(void* const* d_in, const int* in_sizes, int n_in,
                              void* d_out, int out_size, void* d_ws, size_t ws_size,
                              hipStream_t stream) {
  const float* query = (const float*)d_in[0];
  const float* key   = (const float*)d_in[1];
  const int*   mask  = (const int*)d_in[2];
  const float* Wq    = (const float*)d_in[3];
  const float* bq    = (const float*)d_in[4];
  const float* Wk    = (const float*)d_in[5];
  const float* bk    = (const float*)d_in[6];
  const float* Wout  = (const float*)d_in[7];

  float* attn = (float*)d_out;          // [B, H]
  float* aw   = attn + B * H;           // [B, S]

  float* ws = (float*)d_ws;
  float* wv = ws;                       // 512
  float* uv = wv + I;                   // 512
  float* bs = uv + I;                   // 32
  float* mp = bs + B;                   // B*C
  float* lp = mp + B * C;               // B*C
  float* cp = lp + B * C;               // B*C*I  (4 MB)
  float* cf = cp + (long)B * C * I;     // B*I

  prep_wu  <<<2, 256, 0, stream>>>(Wk, Wq, Wout, wv, uv);
  prep_base<<<B, 256, 0, stream>>>(query, uv, bq, bk, Wout, bs);

  size_t shmem = (size_t)(Sc * I + I + Sc + Sc) * sizeof(float);   // ~130.5 KB
  fused_scores<<<B * C, 256, shmem, stream>>>(key, mask, wv, bs, aw, mp, lp, cp);

  combine  <<<B, 256, 0, stream>>>(mp, lp, cp, cf);
  attn_wmma<<<(B / 16) * (H / 16), 32, 0, stream>>>(cf, Wk, bk, attn);
}
